// HybridIDMModel_80238579024143
// MI455X (gfx1250) — matrix operand order repaired and checked by reading, not compiled, not run on hardware
//
#include <hip/hip_runtime.h>
#include <hip/hip_bf16.h>

typedef __attribute__((ext_vector_type(16))) __bf16 v16bf;
typedef __attribute__((ext_vector_type(8)))  __bf16 v8bf;
typedef __attribute__((ext_vector_type(8)))  float  v8f;

#define B_SZ 2048
#define S_SZ 128
#define I_SZ 8
#define H_SZ 256
#define K_SZ 50
#define G4H  1024      // 4*H
#define NT   64        // 16-wide n-tiles across 4H
#define KT0  9         // ceil(288/32): [h0(256) | x(8) | pad(24)]
#define KT1  16        // 512/32:       [h1(256) | h0_new(256)]
#define A0C  296       // padded LDS row stride (288 used) -> conflict-free
#define A1C  520       // padded LDS row stride (512 used) -> conflict-free

#define DT_C     0.1f
#define V_DES_C  12.64798288f
#define T_HW_C   0.50284384f
#define A_MAX_C  0.10033688f
#define B_SAFE_C 4.98937183f
#define S0_C     0.13082412f

// fast transcendentals: v_exp_f32 + v_rcp_f32 (no IEEE div fixup sequence)
__device__ __forceinline__ float sigf(float x) {
  return __builtin_amdgcn_rcpf(1.0f + __expf(-x));
}
__device__ __forceinline__ float tanhfast(float x) {
  float e = __expf(2.0f * x);
  return 1.0f - 2.0f * __builtin_amdgcn_rcpf(e + 1.0f);
}

__device__ __forceinline__ v8f splat8(float v) {
  v8f r;
#pragma unroll
  for (int i = 0; i < 8; ++i) r[i] = v;
  return r;
}

// Opaque zero offset in an SGPR: defeats LICM (so weight-fragment loads stay
// inside the time loop instead of spilling 12KB/lane to scratch) while keeping
// the base pointer's global address space -> global_load_b128, not flat_load.
__device__ __forceinline__ size_t launder_off() {
  size_t z = 0;
  asm volatile("" : "+s"(z));
  return z;
}

__device__ __forceinline__ v16bf loadB(const __bf16* w, int kt, int nt, int lane) {
  return *(const v16bf*)(w + (((size_t)kt * NT + nt) * 32 + lane) * 16);
}

// ---------------- weight pre-pack into WMMA B-fragment layout ----------------
// B frag (16x16x32 bf16): lane<16 -> N=lane,    K = kt*32 + {0..7, 16..23}
//                         lane>=16-> N=lane-16, K = kt*32 + {8..15, 24..31}
__global__ void pack_w0(const float* __restrict__ Whh0, const float* __restrict__ Wih0,
                        __bf16* __restrict__ w0p) {
  const int tile = blockIdx.x;            // kt*NT + nt
  const int kt = tile / NT, nt = tile % NT;
  const int lane = threadIdx.x;
  __bf16* dst = w0p + (((size_t)tile) * 32 + lane) * 16;
  const int n = nt * 16 + (lane & 15);
  const int kbase = kt * 32 + ((lane < 16) ? 0 : 8);
#pragma unroll
  for (int e = 0; e < 16; ++e) {
    const int k = kbase + (e < 8 ? e : e + 8);
    float v = 0.0f;
    if (k < 256)      v = Whh0[(size_t)n * 256 + k];          // vs h0
    else if (k < 264) v = Wih0[(size_t)n * 8 + (k - 256)];    // vs x_t
    dst[e] = (__bf16)v;
  }
}

__global__ void pack_w1(const float* __restrict__ Whh1, const float* __restrict__ Wih1,
                        __bf16* __restrict__ w1p) {
  const int tile = blockIdx.x;
  const int kt = tile / NT, nt = tile % NT;
  const int lane = threadIdx.x;
  __bf16* dst = w1p + (((size_t)tile) * 32 + lane) * 16;
  const int n = nt * 16 + (lane & 15);
  const int kbase = kt * 32 + ((lane < 16) ? 0 : 8);
#pragma unroll
  for (int e = 0; e < 16; ++e) {
    const int k = kbase + (e < 8 ? e : e + 8);
    float v = (k < 256) ? Whh1[(size_t)n * 256 + k]            // vs h1
                        : Wih1[(size_t)n * 256 + (k - 256)];   // vs h0_new
    dst[e] = (__bf16)v;
  }
}

__global__ void bias_sum(const float* __restrict__ bih0, const float* __restrict__ bhh0,
                         const float* __restrict__ bih1, const float* __restrict__ bhh1,
                         float* __restrict__ b0, float* __restrict__ b1) {
  const int i = blockIdx.x * blockDim.x + threadIdx.x;
  if (i < G4H) { b0[i] = bih0[i] + bhh0[i]; b1[i] = bih1[i] + bhh1[i]; }
}

// ---------------- fused persistent 2-layer LSTM ----------------
// 128 blocks x 128 threads (4 waves). Block owns 16 batch rows for all 128 steps.
// Wave w owns h-tiles {4w..4w+3}; c-state lives in registers (C-fragment layout).
__launch_bounds__(128)
__global__ void lstm_fused(const float* __restrict__ x,
                           const __bf16* __restrict__ w0p,
                           const __bf16* __restrict__ w1p,
                           const float* __restrict__ b0s,
                           const float* __restrict__ b1s,
                           float* __restrict__ hfinal) {
  __shared__ __align__(32) __bf16 A0[2][16][A0C];  // [h0(0..255) | x_t(256..263) | zero pad]
  __shared__ __align__(32) __bf16 A1[2][16][A1C];  // [h1(0..255) | h0_new(256..511)]

  const int tid = threadIdx.x;
  const int lane = tid & 31;
  const int w = tid >> 5;
  const int colb = lane & 15;
  const int hiHalf = lane >> 4;                 // 0/1
  const int b0row = blockIdx.x * 16;

  for (int i = tid; i < 2 * 16 * A0C; i += 128) ((__bf16*)A0)[i] = (__bf16)0.0f;
  for (int i = tid; i < 2 * 16 * A1C; i += 128) ((__bf16*)A1)[i] = (__bf16)0.0f;
  {
    const int m = tid >> 3, ii = tid & 7;       // 128 threads = 16 rows x 8 feats
    A0[0][m][256 + ii] = (__bf16)x[((size_t)(b0row + m) * S_SZ + 0) * I_SZ + ii];
  }
  v8f c0r[4], c1r[4];
#pragma unroll
  for (int q = 0; q < 4; ++q) { c0r[q] = splat8(0.0f); c1r[q] = splat8(0.0f); }
  __syncthreads();

  int p = 0;
  for (int t = 0; t < S_SZ; ++t) {
    // per-step opaque weight bases: keep fragment loads inside the t loop
    const __bf16* w0t = w0p + launder_off();
    const __bf16* w1t = w1p + launder_off();

    // ================= layer 0: reads A0[p]; writes A0[1-p] h, A1[p] h0_new =================
    {
      const __bf16* abase = &A0[p][0][0];
#pragma unroll
      for (int q = 0; q < 4; ++q) {
        const int ht = w * 4 + q;
        v8f acc[4];
#pragma unroll
        for (int g = 0; g < 4; ++g) acc[g] = splat8(b0s[g * 256 + ht * 16 + colb]);
        for (int kt = 0; kt < KT0; ++kt) {
          // A frag: lane<16: K {kb..kb+7, kb+16..kb+23}, kb=kt*32 ; lane>=16: kb=kt*32+8
          const __bf16* rp = abase + (size_t)(lane & 15) * A0C + kt * 32 + (hiHalf ? 8 : 0);
          v8bf lo = *(const v8bf*)rp;
          v8bf hi = *(const v8bf*)(rp + 16);
          v16bf a = __builtin_shufflevector(lo, hi, 0,1,2,3,4,5,6,7,8,9,10,11,12,13,14,15);
          v16bf bb[4];
#pragma unroll
          for (int g = 0; g < 4; ++g) bb[g] = loadB(w0t, kt, g * 16 + ht, lane);
#pragma unroll
          for (int g = 0; g < 4; ++g)
            acc[g] = __builtin_amdgcn_wmma_f32_16x16x32_bf16(false, a, false, bb[g],
                                                             (short)0, acc[g], false, false);
        }
        v8f c = c0r[q];
#pragma unroll
        for (int i = 0; i < 8; ++i) {
          const float ig = sigf(acc[0][i]);
          const float fg = sigf(acc[1][i]);
          const float gg = tanhfast(acc[2][i]);
          const float og = sigf(acc[3][i]);
          const float cn = fg * c[i] + ig * gg;
          const float hn = og * tanhfast(cn);
          c[i] = cn;
          const int r = i + (hiHalf << 3);     // C layout: vgpr i, lanes>=16 -> M=i+8
          const int col = ht * 16 + colb;
          const __bf16 hb = (__bf16)hn;
          A0[1 - p][r][col] = hb;              // for next timestep's layer 0
          A1[p][r][256 + col] = hb;            // feeds this timestep's layer 1
        }
        c0r[q] = c;
      }
    }
    __syncthreads();
    if (t + 1 < S_SZ) {                         // prefill x_{t+1}
      const int m = tid >> 3, ii = tid & 7;
      A0[1 - p][m][256 + ii] = (__bf16)x[((size_t)(b0row + m) * S_SZ + (t + 1)) * I_SZ + ii];
    }
    // ================= layer 1: reads A1[p]; writes A1[1-p] h =================
    {
      const __bf16* abase = &A1[p][0][0];
#pragma unroll
      for (int q = 0; q < 4; ++q) {
        const int ht = w * 4 + q;
        v8f acc[4];
#pragma unroll
        for (int g = 0; g < 4; ++g) acc[g] = splat8(b1s[g * 256 + ht * 16 + colb]);
        for (int kt = 0; kt < KT1; ++kt) {
          const __bf16* rp = abase + (size_t)(lane & 15) * A1C + kt * 32 + (hiHalf ? 8 : 0);
          v8bf lo = *(const v8bf*)rp;
          v8bf hi = *(const v8bf*)(rp + 16);
          v16bf a = __builtin_shufflevector(lo, hi, 0,1,2,3,4,5,6,7,8,9,10,11,12,13,14,15);
          v16bf bb[4];
#pragma unroll
          for (int g = 0; g < 4; ++g) bb[g] = loadB(w1t, kt, g * 16 + ht, lane);
#pragma unroll
          for (int g = 0; g < 4; ++g)
            acc[g] = __builtin_amdgcn_wmma_f32_16x16x32_bf16(false, a, false, bb[g],
                                                             (short)0, acc[g], false, false);
        }
        v8f c = c1r[q];
#pragma unroll
        for (int i = 0; i < 8; ++i) {
          const float ig = sigf(acc[0][i]);
          const float fg = sigf(acc[1][i]);
          const float gg = tanhfast(acc[2][i]);
          const float og = sigf(acc[3][i]);
          const float cn = fg * c[i] + ig * gg;
          const float hn = og * tanhfast(cn);
          c[i] = cn;
          const int r = i + (hiHalf << 3);
          const int col = ht * 16 + colb;
          A1[1 - p][r][col] = (__bf16)hn;
          if (t == S_SZ - 1) hfinal[(size_t)(b0row + r) * H_SZ + col] = hn;
        }
        c1r[q] = c;
      }
    }
    __syncthreads();
    p ^= 1;
  }
}

// ---------------- FC head: y_lstm = h_last @ Wfc^T + bfc ----------------
__global__ void fc_head(const float* __restrict__ hf, const float* __restrict__ Wfc,
                        const float* __restrict__ bfc, float* __restrict__ out) {
  const int gid = blockIdx.x * blockDim.x + threadIdx.x;
  if (gid >= B_SZ * K_SZ) return;
  const int b = gid / K_SZ, k = gid % K_SZ;
  const float* hrow = hf + (size_t)b * H_SZ;
  const float* wrow = Wfc + (size_t)k * H_SZ;
  float s = bfc[k];
#pragma unroll 8
  for (int j = 0; j < H_SZ; ++j) s += hrow[j] * wrow[j];
  out[gid] = s;
}

// ---------------- IDM rollout (DELTA == 1.0) ----------------
__global__ void idm_rollout(const float* __restrict__ x, const float* __restrict__ s0,
                            const float* __restrict__ vl, float* __restrict__ out) {
  const int b = blockIdx.x * blockDim.x + threadIdx.x;
  if (b >= B_SZ) return;
  float v = x[((size_t)b * S_SZ + (S_SZ - 1)) * I_SZ + 0];
  float s = s0[b];
  const float vlead = vl[b];
  const float sqrt_ab = __fsqrt_rn(A_MAX_C * B_SAFE_C);
  const float inv2ab = 1.0f / (2.0f * sqrt_ab + 1e-6f);
#pragma unroll
  for (int j = 0; j < K_SZ; ++j) {
    const float dv = vlead - v;
    const float sc = fmaxf(s, 1e-6f);
    const float sstar = fmaxf(S0_C + v * T_HW_C + v * dv * inv2ab, 0.0f);
    const float rat = sstar / sc;
    const float acc = A_MAX_C * (1.0f - (v / V_DES_C) - rat * rat);
    v = fmaxf(v + DT_C * acc, 0.0f);
    s = fmaxf(s + dv * DT_C, 1e-6f);
    out[(size_t)b * K_SZ + j] = v;
  }
}

extern "C" void kernel_launch(void* const* d_in, const int* in_sizes, int n_in,
                              void* d_out, int out_size, void* d_ws, size_t ws_size,
                              hipStream_t stream) {
  (void)in_sizes; (void)n_in; (void)out_size; (void)ws_size;
  const float* x      = (const float*)d_in[0];
  const float* s_safe = (const float*)d_in[1];
  const float* v_lead = (const float*)d_in[2];
  const float* Wih0   = (const float*)d_in[3];
  const float* Whh0   = (const float*)d_in[4];
  const float* bih0   = (const float*)d_in[5];
  const float* bhh0   = (const float*)d_in[6];
  const float* Wih1   = (const float*)d_in[7];
  const float* Whh1   = (const float*)d_in[8];
  const float* bih1   = (const float*)d_in[9];
  const float* bhh1   = (const float*)d_in[10];
  const float* Wfc    = (const float*)d_in[11];
  const float* bfc    = (const float*)d_in[12];
  float* out = (float*)d_out;

  // workspace carve (~3.7 MB total)
  char* ws = (char*)d_ws;
  __bf16* w0p = (__bf16*)ws;                               // KT0*NT*32*16 bf16 = 589,824 B
  __bf16* w1p = (__bf16*)(ws + 589824);                    // KT1*NT*32*16 bf16 = 1,048,576 B
  float*  b0s = (float*)(ws + 589824 + 1048576);           // 4 KB
  float*  b1s = b0s + G4H;                                 // 4 KB
  float*  hfinal = b1s + G4H;                              // B*H f32 = 2 MB

  pack_w0<<<KT0 * NT, 32, 0, stream>>>(Whh0, Wih0, w0p);
  pack_w1<<<KT1 * NT, 32, 0, stream>>>(Whh1, Wih1, w1p);
  bias_sum<<<4, 256, 0, stream>>>(bih0, bhh0, bih1, bhh1, b0s, b1s);
  lstm_fused<<<B_SZ / 16, 128, 0, stream>>>(x, w0p, w1p, b0s, b1s, hfinal);
  fc_head<<<(B_SZ * K_SZ + 255) / 256, 256, 0, stream>>>(hfinal, Wfc, bfc, out);
  idm_rollout<<<(B_SZ + 255) / 256, 256, 0, stream>>>(x, s_safe, v_lead, out + B_SZ * K_SZ);
}
